// LSTM_82652350644534
// MI455X (gfx1250) — matrix operand order repaired and checked by reading, not compiled, not run on hardware
//
#include <hip/hip_runtime.h>

// ---------------------------------------------------------------------------
// LSTM (2-layer, B=32, T=1024, H=512) + FC, bf16 WMMA on gfx1250 (MI455X)
// ---------------------------------------------------------------------------

static constexpr int kB   = 32;
static constexpr int kT   = 1024;
static constexpr int kH   = 512;
static constexpr int kL   = 2;
static constexpr int kOut = 512;
static constexpr int kG   = 4 * kH;   // 2048 gate pre-activations per sample

typedef __bf16 bf16_t;
typedef __attribute__((ext_vector_type(16))) __bf16 v16bf;
typedef __attribute__((ext_vector_type(8)))  __bf16 v8bf;
typedef __attribute__((ext_vector_type(8)))  float  v8f;

#define WMMA_BF16(A, B, C) \
    __builtin_amdgcn_wmma_f32_16x16x32_bf16(false, (A), false, (B), (short)0, (C), false, false)

// ---------------------------------------------------------------------------
// Fragment loaders (wave32 WMMA operands in this code's contiguous k order, 16x16x32 bf16)
//
// A (16x32, MxK): lane holds row m=lane%16.  Element e -> K = k0 + 16*(lane/16) + e
// => one contiguous 32-byte run at (row*stride + k0 + 16*half).
//
// B (32x16, KxN): lane holds col n=lane%16.  Element e -> K = k0 + 16*(lane/16) + e
// => one contiguous 32-byte run at (col*ldb + k0 + 16*half).
// ---------------------------------------------------------------------------
__device__ __forceinline__ v16bf load_a_frag(const bf16_t* __restrict__ p) {
    return *(const v16bf*)(p);
}

__device__ __forceinline__ float sigmoidf_fast(float x) {
    return 1.0f / (1.0f + __expf(-x));
}
__device__ __forceinline__ float tanhf_fast(float x) {
    float e = __expf(2.0f * x);
    return (e - 1.0f) / (e + 1.0f);
}

// ---------------------------------------------------------------------------
// Fragment set for one k-step of the LSTM step GEMMs (4 A + 4 B frags)
// ---------------------------------------------------------------------------
struct LstmFrags {
    v16bf ax0, ax1, ah0, ah1;   // A: input rows (m-tile 0/1), hidden rows (m-tile 0/1)
    v16bf bi0, bi1, bh0, bh1;   // B: Wih n-tile 0/1, Whh n-tile 0/1
};

struct LstmPtrs {
    const bf16_t* ai0; const bf16_t* ai1;
    const bf16_t* ah0; const bf16_t* ah1;
    const bf16_t* wi0; const bf16_t* wi1;
    const bf16_t* wh0; const bf16_t* wh1;
};

__device__ __forceinline__ void lstm_load(LstmFrags& f, const LstmPtrs& p, int k) {
    f.ax0 = load_a_frag(p.ai0 + k);
    f.ax1 = load_a_frag(p.ai1 + k);
    f.ah0 = load_a_frag(p.ah0 + k);
    f.ah1 = load_a_frag(p.ah1 + k);
    f.bi0 = *(const v16bf*)(p.wi0 + k);
    f.bi1 = *(const v16bf*)(p.wi1 + k);
    f.bh0 = *(const v16bf*)(p.wh0 + k);
    f.bh1 = *(const v16bf*)(p.wh1 + k);
}

__device__ __forceinline__ void lstm_mma(const LstmFrags& f, v8f acc[2][2]) {
    acc[0][0] = WMMA_BF16(f.ax0, f.bi0, acc[0][0]);
    acc[0][1] = WMMA_BF16(f.ax0, f.bi1, acc[0][1]);
    acc[1][0] = WMMA_BF16(f.ax1, f.bi0, acc[1][0]);
    acc[1][1] = WMMA_BF16(f.ax1, f.bi1, acc[1][1]);
    acc[0][0] = WMMA_BF16(f.ah0, f.bh0, acc[0][0]);
    acc[0][1] = WMMA_BF16(f.ah0, f.bh1, acc[0][1]);
    acc[1][0] = WMMA_BF16(f.ah1, f.bh0, acc[1][0]);
    acc[1][1] = WMMA_BF16(f.ah1, f.bh1, acc[1][1]);
}

// ---------------------------------------------------------------------------
// One LSTM layer step: gates = inp @ Wih^T + hprev @ Whh^T + bias, then
// pointwise c/h update.  Grid: 8 blocks x 256 threads (8 wave32).
//
// Wave tiling: wave w owns gate = w>>1, unit sub-slice uh = (w&1)*32,
// BOTH batch m-tiles, 2 N-tiles: 8 loads + 8 WMMAs per k-step, no
// duplicated weight fragments within a block.
//
// Ping-pong software pipeline (manual 2x unroll, two fixed fragment sets):
// loads for set f1 overlap WMMAs on f0 and vice versa -> no rotation moves,
// no WMMA->VALU WAR hazard NOPs, partial s_wait_loadcnt only.
// ---------------------------------------------------------------------------
__global__ __launch_bounds__(256, 1) void lstm_step_kernel(
    const bf16_t* __restrict__ inp,  long inp_stride,   // row b at inp + b*inp_stride
    const bf16_t* __restrict__ hprev,                    // [kB][kH]
    const bf16_t* __restrict__ Wih,                      // [4H][H] row-major bf16
    const bf16_t* __restrict__ Whh,                      // [4H][H]
    const float*  __restrict__ bias,                     // [4H] (b_ih + b_hh)
    float*        __restrict__ c,                        // [kB][kH] this layer's cell state
    bf16_t*       __restrict__ hout,                     // [kB][kH] new h (bf16)
    bf16_t*       __restrict__ h1out, long h1_stride,    // row b at h1out + b*h1_stride
    int write_h1)
{
    __shared__ float gbuf[4 * kB * 64];   // [gate][batch][unit] fp32, 32 KB

    const int tid  = threadIdx.x;
    const int lane = tid & 31;
    const int wid  = tid >> 5;
    const int half = lane >> 4;
    const int l16  = lane & 15;

    const int u0   = blockIdx.x * 64;     // hidden-unit slice [u0, u0+64)
    const int gate = wid >> 1;            // 0=i 1=f 2=g 3=o
    const int uh   = (wid & 1) * 32;      // unit sub-slice within the 64

    LstmPtrs p;
    p.ai0 = inp   + (size_t)(l16)      * inp_stride + 16 * half;
    p.ai1 = inp   + (size_t)(16 + l16) * inp_stride + 16 * half;
    p.ah0 = hprev + (size_t)(l16)      * kH         + 16 * half;
    p.ah1 = hprev + (size_t)(16 + l16) * kH         + 16 * half;
    const size_t ng0 = (size_t)(gate * kH + u0 + uh + 0  + l16);
    const size_t ng1 = (size_t)(gate * kH + u0 + uh + 16 + l16);
    p.wi0 = Wih + ng0 * kH + 16 * half;
    p.wi1 = Wih + ng1 * kH + 16 * half;
    p.wh0 = Whh + ng0 * kH + 16 * half;
    p.wh1 = Whh + ng1 * kH + 16 * half;

    v8f acc[2][2] = {};   // [m-tile][n-tile]

    LstmFrags f0, f1;
    lstm_load(f0, p, 0);                    // prologue: k-step 0
#pragma unroll 1
    for (int k = 0; k + 64 < kH; k += 64) { // 7 iterations (k-steps 1..14)
        lstm_load(f1, p, k + 32);
        lstm_mma(f0, acc);
        lstm_load(f0, p, k + 64);
        lstm_mma(f1, acc);
    }
    lstm_load(f1, p, kH - 32);              // epilogue: k-steps 14 + 15
    lstm_mma(f0, acc);
    lstm_mma(f1, acc);

    // ---- epilogue: + bias, spill gates to LDS ----
#pragma unroll
    for (int mt = 0; mt < 2; ++mt) {
#pragma unroll
        for (int nt = 0; nt < 2; ++nt) {
            const int u  = uh + nt * 16 + l16;            // unit within slice, 0..63
            const float bb = bias[gate * kH + u0 + u];
#pragma unroll
            for (int r = 0; r < 8; ++r) {
                const int br = mt * 16 + r + 8 * half;    // batch row 0..31
                gbuf[(gate * kB + br) * 64 + u] = acc[mt][nt][r] + bb;
            }
        }
    }
    __syncthreads();

    // ---- fused pointwise LSTM cell update (32*64 = 2048 elems / block) ----
#pragma unroll
    for (int q = 0; q < 8; ++q) {
        const int flat = tid + q * 256;
        const int b = flat >> 6;
        const int u = flat & 63;
        const float gi = gbuf[(0 * kB + b) * 64 + u];
        const float gf = gbuf[(1 * kB + b) * 64 + u];
        const float gg = gbuf[(2 * kB + b) * 64 + u];
        const float go = gbuf[(3 * kB + b) * 64 + u];
        const size_t ci = (size_t)b * kH + u0 + u;
        const float c_old = c[ci];
        const float c_new = sigmoidf_fast(gf) * c_old + sigmoidf_fast(gi) * tanhf_fast(gg);
        const float h_new = sigmoidf_fast(go) * tanhf_fast(c_new);
        c[ci]    = c_new;
        hout[ci] = (bf16_t)h_new;
        if (write_h1)
            h1out[(size_t)b * h1_stride + u0 + u] = (bf16_t)h_new;
    }
}

// ---------------------------------------------------------------------------
// Final FC: out[r, o] = sum_k h1[r, k] * Wfc[o, k] + bfc[o]
// M = B*T = 32768, N = 512, K = 512.  Block tile 128x64, wave tile 16x64.
// Same ping-pong pipeline (1 A + 4 B frags per set).
// ---------------------------------------------------------------------------
struct FcFrags { v16bf a, b0, b1, b2, b3; };
struct FcPtrs {
    const bf16_t* ar;
    const bf16_t* w0; const bf16_t* w1; const bf16_t* w2; const bf16_t* w3;
};

__device__ __forceinline__ void fc_load(FcFrags& f, const FcPtrs& p, int k) {
    f.a  = load_a_frag(p.ar + k);
    f.b0 = *(const v16bf*)(p.w0 + k);
    f.b1 = *(const v16bf*)(p.w1 + k);
    f.b2 = *(const v16bf*)(p.w2 + k);
    f.b3 = *(const v16bf*)(p.w3 + k);
}

__device__ __forceinline__ void fc_mma(const FcFrags& f, v8f acc[4]) {
    acc[0] = WMMA_BF16(f.a, f.b0, acc[0]);
    acc[1] = WMMA_BF16(f.a, f.b1, acc[1]);
    acc[2] = WMMA_BF16(f.a, f.b2, acc[2]);
    acc[3] = WMMA_BF16(f.a, f.b3, acc[3]);
}

__global__ __launch_bounds__(256, 1) void fc_kernel(
    const bf16_t* __restrict__ h1,    // [B*T][H] bf16, row r = b*T + t
    const bf16_t* __restrict__ Wfc,   // [OUT][H] bf16
    const float*  __restrict__ bfc,   // [OUT]
    float*        __restrict__ out)   // [B*T][OUT]
{
    const int tid  = threadIdx.x;
    const int lane = tid & 31;
    const int wid  = tid >> 5;
    const int half = lane >> 4;
    const int l16  = lane & 15;

    const int m0 = blockIdx.x * 128 + wid * 16;  // gridDim.x = 256
    const int n0 = blockIdx.y * 64;              // gridDim.y = 8

    FcPtrs p;
    p.ar = h1 + (size_t)(m0 + l16) * kH + 16 * half;
    p.w0 = Wfc + (size_t)(n0 + 0  + l16) * kH + 16 * half;
    p.w1 = Wfc + (size_t)(n0 + 16 + l16) * kH + 16 * half;
    p.w2 = Wfc + (size_t)(n0 + 32 + l16) * kH + 16 * half;
    p.w3 = Wfc + (size_t)(n0 + 48 + l16) * kH + 16 * half;

    v8f acc[4] = {};

    FcFrags f0, f1;
    fc_load(f0, p, 0);
#pragma unroll 1
    for (int k = 0; k + 64 < kH; k += 64) {
        fc_load(f1, p, k + 32);
        fc_mma(f0, acc);
        fc_load(f0, p, k + 64);
        fc_mma(f1, acc);
    }
    fc_load(f1, p, kH - 32);
    fc_mma(f0, acc);
    fc_mma(f1, acc);

#pragma unroll
    for (int nt = 0; nt < 4; ++nt) {
        const int col = n0 + nt * 16 + l16;
        const float bb = bfc[col];
#pragma unroll
        for (int r = 0; r < 8; ++r) {
            const int row = m0 + r + 8 * half;
            out[(size_t)row * kOut + col] = acc[nt][r] + bb;
        }
    }
}

// ---------------------------------------------------------------------------
// Setup kernels: fp32 -> bf16 conversion, bias combine, state zeroing
// ---------------------------------------------------------------------------
__global__ void cvt_f32_bf16_kernel(const float* __restrict__ in,
                                    bf16_t* __restrict__ out, long n) {
    long i = (long)blockIdx.x * blockDim.x + threadIdx.x;
    long stride = (long)gridDim.x * blockDim.x;
    for (; i < n; i += stride) out[i] = (bf16_t)in[i];
}

__global__ void bias_combine_kernel(const float* __restrict__ a,
                                    const float* __restrict__ b,
                                    float* __restrict__ o, long n) {
    long i = (long)blockIdx.x * blockDim.x + threadIdx.x;
    long stride = (long)gridDim.x * blockDim.x;
    for (; i < n; i += stride) o[i] = a[i] + b[i];
}

__global__ void zero_state_kernel(bf16_t* __restrict__ h, long nh,
                                  float* __restrict__ c, long nc) {
    long i = (long)blockIdx.x * blockDim.x + threadIdx.x;
    long stride = (long)gridDim.x * blockDim.x;
    for (long j = i; j < nh; j += stride) h[j] = (bf16_t)0.0f;
    for (long j = i; j < nc; j += stride) c[j] = 0.0f;
}

// ---------------------------------------------------------------------------
// Host launcher (graph-capture safe: no allocs, all work on `stream`)
// ---------------------------------------------------------------------------
extern "C" void kernel_launch(void* const* d_in, const int* in_sizes, int n_in,
                              void* d_out, int out_size, void* d_ws, size_t ws_size,
                              hipStream_t stream) {
    (void)in_sizes; (void)n_in; (void)out_size; (void)ws_size;

    const float* x    = (const float*)d_in[0];  // [B][T][H]
    const float* W_ih = (const float*)d_in[1];  // [L][4H][H]
    const float* W_hh = (const float*)d_in[2];  // [L][4H][H]
    const float* b_ih = (const float*)d_in[3];  // [L][4H]
    const float* b_hh = (const float*)d_in[4];  // [L][4H]
    const float* W_fc = (const float*)d_in[5];  // [OUT][H]
    const float* b_fc = (const float*)d_in[6];  // [OUT]
    float* out = (float*)d_out;                 // [B*T][OUT]

    // ---- workspace carve-up (all offsets 256B aligned by construction) ----
    char* ws = (char*)d_ws;
    size_t off = 0;
    auto take = [&](size_t bytes) {
        char* p = ws + off;
        off += (bytes + 255) & ~(size_t)255;
        return p;
    };
    bf16_t* xbf   = (bf16_t*)take((size_t)kB * kT * kH * sizeof(bf16_t));   // 32 MB
    bf16_t* Wihb  = (bf16_t*)take((size_t)kL * kG * kH * sizeof(bf16_t));   // 4 MB
    bf16_t* Whhb  = (bf16_t*)take((size_t)kL * kG * kH * sizeof(bf16_t));   // 4 MB
    bf16_t* Wfcb  = (bf16_t*)take((size_t)kOut * kH * sizeof(bf16_t));      // 0.5 MB
    float*  biasb = (float*) take((size_t)kL * kG * sizeof(float));         // 16 KB
    bf16_t* hbuf  = (bf16_t*)take((size_t)2 * kL * kB * kH * sizeof(bf16_t)); // ping-pong h
    float*  cbuf  = (float*) take((size_t)kL * kB * kH * sizeof(float));
    bf16_t* h1all = (bf16_t*)take((size_t)kB * kT * kH * sizeof(bf16_t));   // 32 MB

    // ---- phase 0: convert / combine / zero ----
    const long nX = (long)kB * kT * kH;
    const long nW = (long)kL * kG * kH;
    cvt_f32_bf16_kernel<<<2048, 256, 0, stream>>>(x,    xbf,  nX);
    cvt_f32_bf16_kernel<<<1024, 256, 0, stream>>>(W_ih, Wihb, nW);
    cvt_f32_bf16_kernel<<<1024, 256, 0, stream>>>(W_hh, Whhb, nW);
    cvt_f32_bf16_kernel<<<256,  256, 0, stream>>>(W_fc, Wfcb, (long)kOut * kH);
    bias_combine_kernel<<<16,   256, 0, stream>>>(b_ih, b_hh, biasb, (long)kL * kG);
    zero_state_kernel<<<256, 256, 0, stream>>>(hbuf, (long)2 * kL * kB * kH,
                                               cbuf, (long)kL * kB * kH);

    // ---- phase 1: recurrence (2 layer-kernels per timestep) ----
    const size_t layerW = (size_t)kG * kH;     // bf16 elems per layer weight
    const size_t hSlab  = (size_t)kB * kH;     // elems per [B][H] state slab
    for (int t = 0; t < kT; ++t) {
        const int par = t & 1;
        bf16_t* h0_prev = hbuf + ((size_t)par       * kL + 0) * hSlab;
        bf16_t* h0_new  = hbuf + ((size_t)(par ^ 1) * kL + 0) * hSlab;
        bf16_t* h1_prev = hbuf + ((size_t)par       * kL + 1) * hSlab;
        bf16_t* h1_new  = hbuf + ((size_t)(par ^ 1) * kL + 1) * hSlab;

        // layer 0: input = x[:, t, :]  (row b at xbf + (b*T + t)*H)
        lstm_step_kernel<<<8, 256, 0, stream>>>(
            xbf + (size_t)t * kH, (long)kT * kH,
            h0_prev,
            Wihb, Whhb, biasb,
            cbuf, h0_new,
            nullptr, 0, 0);

        // layer 1: input = layer-0 h at this timestep
        lstm_step_kernel<<<8, 256, 0, stream>>>(
            h0_new, (long)kH,
            h1_prev,
            Wihb + layerW, Whhb + layerW, biasb + kG,
            cbuf + hSlab, h1_new,
            h1all + (size_t)t * kH, (long)kT * kH, 1);
    }

    // ---- phase 2: final FC over all (b, t) rows ----
    fc_kernel<<<dim3((kB * kT) / 128, kOut / 64), 256, 0, stream>>>(
        h1all, Wfcb, b_fc, out);
}
